// Sequential2D_10419590660264
// MI455X (gfx1250) — compile-verified, matrix operand order checked by reading, not run on hardware
//
#include <hip/hip_runtime.h>
#include <hip/hip_bf16.h>

// Problem constants (from the reference)
#define NBLK   16
#define BATCH  16384
#define BLK    128

// Tiling
#define TILE_M 128       // M rows per workgroup
#define TILE_N 128       // N cols per workgroup (= one output block)
#define KC     32        // K chunk per LDS stage
#define LDF    36        // f32 LDS row stride (32 + 4 pad), 144B, 16B-aligned
#define LDB    40        // bf16 LDS row stride (32 + 8 pad), 80B, 16B-aligned

#define SF_BYTES (TILE_M * LDF * 4)   // 18432 B per f32 staging buffer

typedef __bf16 v16bf __attribute__((ext_vector_type(16)));
typedef __bf16 v8bf  __attribute__((ext_vector_type(8)));
typedef __bf16 v4bf  __attribute__((ext_vector_type(4)));
typedef float  v8f   __attribute__((ext_vector_type(8)));
typedef int    v4i   __attribute__((vector_size(16)));  // int4, matches builtin param

typedef __attribute__((address_space(1))) v4i as1_v4i;  // global int4
typedef __attribute__((address_space(3))) v4i as3_v4i;  // LDS int4

// generic -> addrspace pointers via integer round-trip (LDS generic addr low
// 32 bits == LDS offset per CDNA5 aperture rules)
static __device__ __forceinline__ as3_v4i* lds_as3(void* p) {
  return (as3_v4i*)(uint32_t)(uintptr_t)p;
}
static __device__ __forceinline__ as1_v4i* gbl_as1(const void* p) {
  return (as1_v4i*)(uintptr_t)p;
}

#if defined(__has_builtin)
#if __has_builtin(__builtin_amdgcn_global_load_async_to_lds_b128)
#define HAVE_ASYNC_LDS 1
#endif
#endif

static __device__ __forceinline__ void async_copy_b128(void* lds, const void* g) {
#if defined(HAVE_ASYNC_LDS)
  // (global src int4*, LDS dst int4*, imm offset, imm cpol)
  __builtin_amdgcn_global_load_async_to_lds_b128(gbl_as1(g), lds_as3(lds), 0, 0);
#else
  *(float4*)lds = *(const float4*)g;
#endif
}

#if defined(HAVE_ASYNC_LDS)
#if __has_builtin(__builtin_amdgcn_s_wait_asynccnt)
#define WAIT_ASYNC(n) __builtin_amdgcn_s_wait_asynccnt(n)
#else
#define WAIT_ASYNC(n) asm volatile("s_wait_asynccnt %0" ::"i"(n) : "memory")
#endif
#else
#define WAIT_ASYNC(n) do {} while (0)
#endif

// A/B fragment loader for 16-bit WMMA operands (wave32).
// p points at &row[half*8] where half = lane>>4.
// Elements 0..7  = K = half*8 + 0..7
// Elements 8..15 = K = 16 + half*8 + 0..7   -> two ds_load_b128.
static __device__ __forceinline__ v16bf load_frag(const __bf16* p) {
  v8bf a = *(const v8bf*)p;
  v8bf b = *(const v8bf*)(p + 16);
  v16bf r;
#pragma unroll
  for (int i = 0; i < 8; ++i) { r[i] = a[i]; r[8 + i] = b[i]; }
  return r;
}

__global__ __launch_bounds__(256) void seq2d_wmma_kernel(
    const float* __restrict__ X, const float* __restrict__ W,
    const float* __restrict__ B, float* __restrict__ Y) {
  // Double-buffered f32 staging (async DMA targets) + single bf16 hi/lo
  // operand buffers. Total static LDS ~= 2*18432*2 + 4*10240 + 512 = 115 KB
  // -> 2 workgroups per 320 KB WGP, so convert/compute of one WG overlaps
  // the other's.
  __shared__ __align__(16) unsigned char smXf[2][SF_BYTES];
  __shared__ __align__(16) unsigned char smWf[2][SF_BYTES];
  __shared__ __align__(16) __bf16 sXh[TILE_M * LDB];
  __shared__ __align__(16) __bf16 sXl[TILE_M * LDB];
  __shared__ __align__(16) __bf16 sWh[TILE_N * LDB];
  __shared__ __align__(16) __bf16 sWl[TILE_N * LDB];
  __shared__ float sBias[BLK];

  const int tid  = threadIdx.x;
  const int lane = tid & 31;
  const int wave = tid >> 5;
  const int wm   = wave & 1;   // 2 waves across M: 64 rows each
  const int wn   = wave >> 1;  // 4 waves across N: 32 cols each

  // L2-reuse-friendly mapping: o (output block) is the FAST grid dimension,
  // so all 16 o-WGs sharing one M-tile's X slice (and all of W, 8.4 MB) are
  // concurrently resident -> X and W stream from HBM once and hit in the
  // 192 MB L2 thereafter (~277 MB total HBM traffic, the roofline minimum).
  const int o    = blockIdx.x; // output block row 0..15
  const int p    = o & 1;
  const int m0   = blockIdx.y * TILE_M;

  const int nchunks = (NBLK / 2) * BLK / KC;  // 32

  // Issue one chunk's async loads: 128x32 f32 tiles of X and W
  // (1024 b128 each -> 4 per thread per matrix, 8 ASYNCcnt per wave).
  auto issue_chunk = [&](int chunk) {
    const int j   = chunk >> 2;         // 4 chunks of 32 per 128-wide j block
    const int kk0 = (chunk & 3) * KC;
    const int jb  = 2 * j + p;
    const float* gx = X + (((size_t)jb * BATCH) + m0) * BLK + kk0;
    const float* gw = W + ((((size_t)o * NBLK) + jb) * (size_t)BLK) * BLK + kk0;
    float* bXf = (float*)smXf[chunk & 1];
    float* bWf = (float*)smWf[chunk & 1];
#pragma unroll
    for (int it = 0; it < 4; ++it) {
      int idx = it * 256 + tid;          // 0..1023
      int row = idx >> 3;                // 8 x b128 per row
      int c4  = (idx & 7) * 4;
      async_copy_b128(&bXf[row * LDF + c4], gx + (size_t)row * BLK + c4);
      async_copy_b128(&bWf[row * LDF + c4], gw + (size_t)row * BLK + c4);
    }
  };

  // bias[f] = sum_j b[o, 2j+p, f]
  if (tid < BLK) {
    float s = 0.f;
#pragma unroll
    for (int j = 0; j < NBLK / 2; ++j)
      s += B[((size_t)o * NBLK + (2 * j + p)) * BLK + tid];
    sBias[tid] = s;
  }

  v8f acc[4][2] = {};

  issue_chunk(0);  // prologue prefetch

  for (int chunk = 0; chunk < nchunks; ++chunk) {
    // Prefetch next chunk, then wait for the current one (async loads
    // complete in order, so <=8 outstanding means chunk's 8 have landed).
    if (chunk + 1 < nchunks) {
      issue_chunk(chunk + 1);
      WAIT_ASYNC(8);
    } else {
      WAIT_ASYNC(0);
    }
    __syncthreads();

    // f32 -> bf16 hi/lo split into the operand buffers (b64 packed stores)
    const float* bXf = (const float*)smXf[chunk & 1];
    const float* bWf = (const float*)smWf[chunk & 1];
#pragma unroll
    for (int it = 0; it < 4; ++it) {
      int idx = it * 256 + tid;
      int row = idx >> 3;
      int c4  = (idx & 7) * 4;
      float4 xv = *(const float4*)&bXf[row * LDF + c4];
      float4 wv = *(const float4*)&bWf[row * LDF + c4];
      v4bf xh, xl, wh, wl;
#pragma unroll
      for (int e = 0; e < 4; ++e) {
        float x = ((const float*)&xv)[e];
        __bf16 h = (__bf16)x;
        xh[e] = h;
        xl[e] = (__bf16)(x - (float)h);
        float w = ((const float*)&wv)[e];
        __bf16 hw = (__bf16)w;
        wh[e] = hw;
        wl[e] = (__bf16)(w - (float)hw);
      }
      *(v4bf*)&sXh[row * LDB + c4] = xh;
      *(v4bf*)&sXl[row * LDB + c4] = xl;
      *(v4bf*)&sWh[row * LDB + c4] = wh;
      *(v4bf*)&sWl[row * LDB + c4] = wl;
    }
    __syncthreads();

    // Compute: one K=32 step, 8 tiles/wave, 3 WMMAs per tile (bf16x3 ~ f32)
    const int half = lane >> 4;
    const int r16  = lane & 15;
    v16bf Ah[4], Al[4], Bh[2], Bl[2];
#pragma unroll
    for (int mt = 0; mt < 4; ++mt) {
      int row = wm * 64 + mt * 16 + r16;
      Ah[mt] = load_frag(&sXh[row * LDB + half * 8]);
      Al[mt] = load_frag(&sXl[row * LDB + half * 8]);
    }
#pragma unroll
    for (int nt = 0; nt < 2; ++nt) {
      int rown = wn * 32 + nt * 16 + r16;
      Bh[nt] = load_frag(&sWh[rown * LDB + half * 8]);
      Bl[nt] = load_frag(&sWl[rown * LDB + half * 8]);
    }
#pragma unroll
    for (int mt = 0; mt < 4; ++mt) {
#pragma unroll
      for (int nt = 0; nt < 2; ++nt) {
        v8f c = acc[mt][nt];
        c = __builtin_amdgcn_wmma_f32_16x16x32_bf16(false, Ah[mt], false, Bh[nt],
                                                    (short)0, c, false, false);
        c = __builtin_amdgcn_wmma_f32_16x16x32_bf16(false, Ah[mt], false, Bl[nt],
                                                    (short)0, c, false, false);
        c = __builtin_amdgcn_wmma_f32_16x16x32_bf16(false, Al[mt], false, Bh[nt],
                                                    (short)0, c, false, false);
        acc[mt][nt] = c;
      }
    }
    __syncthreads();  // compute done before next convert overwrites bf16 bufs
  }

  // Epilogue: C/D layout -> VGPR r: lanes 0-15 hold M=r, lanes 16-31 M=8+r
  const int r16   = lane & 15;
  const int mhalf = (lane >> 4) * 8;
#pragma unroll
  for (int nt = 0; nt < 2; ++nt) {
    int n = wn * 32 + nt * 16 + r16;
    float bias = sBias[n];
#pragma unroll
    for (int mt = 0; mt < 4; ++mt) {
#pragma unroll
      for (int r = 0; r < 8; ++r) {
        int m = m0 + wm * 64 + mt * 16 + mhalf + r;
        Y[((size_t)o * BATCH + m) * BLK + n] = acc[mt][nt][r] + bias;
      }
    }
  }
}

extern "C" void kernel_launch(void* const* d_in, const int* in_sizes, int n_in,
                              void* d_out, int out_size, void* d_ws, size_t ws_size,
                              hipStream_t stream) {
  const float* X = (const float*)d_in[0];
  const float* W = (const float*)d_in[1];
  const float* B = (const float*)d_in[2];
  float* Y = (float*)d_out;
  (void)in_sizes; (void)n_in; (void)out_size; (void)d_ws; (void)ws_size;

  dim3 grid(NBLK, BATCH / TILE_M, 1);  // (16, 128): o fast, M-tile slow
  dim3 block(256, 1, 1);               // 8 wave32s
  seq2d_wmma_kernel<<<grid, block, 0, stream>>>(X, W, B, Y);
}
